// CriterionLP_75909251990005
// MI455X (gfx1250) — matrix operand order repaired and checked by reading, not compiled, not run on hardware
//
#include <hip/hip_runtime.h>
#include <hip/hip_bf16.h>

typedef _Float16 v16h __attribute__((ext_vector_type(16)));
typedef _Float16 v8h  __attribute__((ext_vector_type(8)));
typedef float    v8f  __attribute__((ext_vector_type(8)));

#define B_ROWS   4096
#define C_DIM    128
#define TOPK_    8
#define K_INST_  16
#define P_IDS_   256                 // B_ROWS / K_INST
#define BLK_COLS 128                 // K_INST * TOPK
#define N_COLS   (B_ROWS * TOPK_)    // 32768
#define INV_T    20.0f               // 1 / 0.05
#define NWAVES   4
#define FLT_BIG  3.402823466e38f
#define ROW_H    136                 // 128 halves + 8 pad (16B) -> distinct LDS banks

// ---------------------------------------------------------------------------
// 1) Convert f32 inputs -> f16 in workspace (both arrays are L2-resident).
// ---------------------------------------------------------------------------
__global__ void cvt_f32_to_f16_kernel(const float* __restrict__ feats,
                                      const float* __restrict__ feats_s,
                                      _Float16* __restrict__ feats_h,
                                      _Float16* __restrict__ fs_h) {
    const int n1 = (B_ROWS * C_DIM) / 4;         // feats, in float4 units
    const int n2 = (N_COLS * C_DIM) / 4;         // feats_s, in float4 units
    int i = blockIdx.x * blockDim.x + threadIdx.x;
    if (i >= n1 + n2) return;
    const float4*  src;
    _Float16*      dst;
    int j;
    if (i < n1) { src = (const float4*)feats;   dst = feats_h; j = i; }
    else        { src = (const float4*)feats_s; dst = fs_h;    j = i - n1; }
    float4 v = src[j];
    dst[4 * j + 0] = (_Float16)v.x;
    dst[4 * j + 1] = (_Float16)v.y;
    dst[4 * j + 2] = (_Float16)v.z;
    dst[4 * j + 3] = (_Float16)v.w;
}

// ---------------------------------------------------------------------------
// Async-DMA one 32KB identity block (128 fs rows x 256B) into an LDS buffer
// with 16B/row padding. 128 threads x 16B per op -> 8 rows/op, 16 ops total.
// Tracked on ASYNCcnt; each wave issues 16 ops per block.
// ---------------------------------------------------------------------------
__device__ __forceinline__ void stage_block_async(const _Float16* __restrict__ fs_h,
                                                  int p,
                                                  _Float16 (*buf)[ROW_H],
                                                  int tid) {
    const int rsub  = tid >> 4;          // 0..7 : row within a pass
    const int chunk = tid & 15;          // 16B chunk within the 256B row
    const _Float16* g0 = fs_h + (size_t)p * BLK_COLS * C_DIM;
    #pragma unroll
    for (int i = 0; i < 16; ++i) {
        const int row = i * 8 + rsub;
        unsigned long long gaddr =
            (unsigned long long)(uintptr_t)(g0 + (size_t)row * C_DIM + chunk * 8);
        unsigned int laddr =
            (unsigned int)(uintptr_t)(&buf[row][chunk * 8]);
        asm volatile("global_load_async_to_lds_b128 %0, %1, off"
                     :: "v"(laddr), "v"(gaddr) : "memory");
    }
}

// ---------------------------------------------------------------------------
// Wave computes its 2 fs-row tiles of the staged block (A operand from LDS)
// against the persistent anchor fragments (B operand), then reduces over fs
// rows (the M dimension = the 8 in-lane accumulator components + the two
// lane-halves). Returns, in every lane, the min (OWN) / max over the wave's
// 32 fs rows for anchor n = lane%16.
// ---------------------------------------------------------------------------
template <bool OWN>
__device__ __forceinline__ float tiles_from_lds(const v16h b[4],
                                                const _Float16 (*buf)[ROW_H],
                                                int w, int hl, int lm) {
    float red = OWN ? FLT_BIG : -FLT_BIG;
    #pragma unroll
    for (int tt = 0; tt < 2; ++tt) {
        const int t = 2 * w + tt;
        // A-fragment: fs row = t*16 + lm; contiguous K order (matches B):
        // elem e -> K = kk + hl*16 + e
        const _Float16* base = &buf[t * 16 + lm][0];
        v8f c = {0.f, 0.f, 0.f, 0.f, 0.f, 0.f, 0.f, 0.f};
        #pragma unroll
        for (int k = 0; k < 4; ++k) {
            const int kk = k * 32;
            v8h lo = *(const v8h*)(base + kk +     hl * 16);   // ds_load_b128
            v8h hi = *(const v8h*)(base + kk + 8 + hl * 16);   // ds_load_b128
            v16h af;
            #pragma unroll
            for (int e = 0; e < 8; ++e) { af[e] = lo[e]; af[e + 8] = hi[e]; }
            c = __builtin_amdgcn_wmma_f32_16x16x32_f16(
                    false, af, false, b[k], (short)0, c, false, false);
        }
        // D(m,n): lane n holds m = hl*8 + v in component v -> in-lane reduce.
        #pragma unroll
        for (int v = 0; v < 8; ++v)
            red = OWN ? fminf(red, c[v]) : fmaxf(red, c[v]);
    }
    // Fold the two lane-halves (m 0..7 vs m 8..15): one full-wave xor-16.
    float y = __shfl_xor(red, 16, 32);
    return OWN ? fminf(red, y) : fmaxf(red, y);
}

// ---------------------------------------------------------------------------
// 2) Fused GEMM + per-identity-block min/max reduction + loss.
//    Grid: 256 WGs x 128 threads. WG b: anchors [16b,16b+16). All 4 waves
//    cooperate on one column block at a time; blocks double-buffered in LDS
//    via async-to-LDS DMA.
// ---------------------------------------------------------------------------
__global__ __launch_bounds__(NWAVES * 32)
void lp_loss_kernel(const _Float16* __restrict__ feats_h,
                    const _Float16* __restrict__ fs_h,
                    const int* __restrict__ labels,
                    float* __restrict__ part) {
    __shared__ _Float16 sbuf[2][BLK_COLS][ROW_H] __attribute__((aligned(64)));
    __shared__ float    s_part[NWAVES][16];

    const int tid    = threadIdx.x;
    const int w      = tid >> 5;           // wave id 0..3
    const int l      = tid & 31;           // lane id
    const int hl     = l >> 4;             // lane half
    const int lm     = l & 15;             // lane within half (= anchor index)
    const int b_base = blockIdx.x * 16;
    const int p_own  = __builtin_amdgcn_readfirstlane(labels[b_base]); // SGPR

    // ---- Persistent B fragments: anchors as columns. Lane holds column
    // n = lm -> feats row (b_base+lm), K = kk + hl*16 .. +15 contiguous. ----
    v16h b[4];
    {
        const _Float16* brow = feats_h + (size_t)(b_base + lm) * C_DIM + hl * 16;
        #pragma unroll
        for (int k = 0; k < 4; ++k)
            b[k] = *(const v16h*)(brow + k * 32);
    }

    // Per-anchor accumulators live in lanes 0..15 of wave 0 (anchor r = lane).
    float neg_r = 0.0f;
    float pos_r = FLT_BIG;

    // Prologue: stage block 0 into buffer 0.
    stage_block_async(fs_h, 0, sbuf[0], tid);

    for (int p = 0; p < P_IDS_; ++p) {
        const int  cur = p & 1;
        const bool own = (p == p_own);     // scalar (SGPR) comparison

        // Kick off DMA for the next block into the other buffer.
        stage_block_async(fs_h, (p + 1) & (P_IDS_ - 1), sbuf[cur ^ 1], tid);
        // Wait for THIS block's 16 copies (the 16 just issued stay in flight),
        // then make all waves' copies visible.
        asm volatile("s_wait_asynccnt 16" ::: "memory");
        __syncthreads();                   // barrier A: buf[cur] ready

        float red = own ? tiles_from_lds<true >(b, sbuf[cur], w, hl, lm)
                        : tiles_from_lds<false>(b, sbuf[cur], w, hl, lm);

        if (l < 16) s_part[w][l] = red;    // lane n -> anchor n, one store
        __syncthreads();                   // barrier B: partials visible,
                                           // all reads of buf[cur] retired
        if (tid < 16) {                    // wave 0 combines 4 wave-partials
            float m = s_part[0][tid];
            #pragma unroll
            for (int ww = 1; ww < NWAVES; ++ww)
                m = own ? fminf(m, s_part[ww][tid]) : fmaxf(m, s_part[ww][tid]);
            if (own) pos_r = m;
            else     neg_r += __expf(m * INV_T);  // exp at block granularity
        }
    }

    // ---- Per-anchor loss (wave 0, lanes 0..15 hold anchors 0..15). ----
    if (tid < 16) {
        float pos  = __expf(pos_r * INV_T);
        float loss = -__logf(pos / (pos + neg_r + 1e-6f) + 1e-6f);
        #pragma unroll
        for (int off = 8; off >= 1; off >>= 1)
            loss += __shfl_xor(loss, off, 16);
        if (tid == 0) part[blockIdx.x] = loss;   // sum of 16 anchor losses
    }
}

// ---------------------------------------------------------------------------
// 3) Deterministic final reduction over the 256 per-WG partials -> mean.
// ---------------------------------------------------------------------------
__global__ void finalize_kernel(const float* __restrict__ part,
                                float* __restrict__ out) {
    __shared__ float s[P_IDS_];
    s[threadIdx.x] = part[threadIdx.x];
    __syncthreads();
    #pragma unroll
    for (int off = P_IDS_ / 2; off >= 1; off >>= 1) {
        if ((int)threadIdx.x < off) s[threadIdx.x] += s[threadIdx.x + off];
        __syncthreads();
    }
    if (threadIdx.x == 0) out[0] = s[0] * (1.0f / B_ROWS);
}

// ---------------------------------------------------------------------------
extern "C" void kernel_launch(void* const* d_in, const int* in_sizes, int n_in,
                              void* d_out, int out_size, void* d_ws, size_t ws_size,
                              hipStream_t stream) {
    const float* feats   = (const float*)d_in[0];   // [4096, 128]
    const float* feats_s = (const float*)d_in[1];   // [4096, 8, 128]
    const int*   labels  = (const int*)d_in[2];     // [4096]
    float* out = (float*)d_out;

    _Float16* feats_h = (_Float16*)d_ws;                          // 1 MB
    _Float16* fs_h    = feats_h + (size_t)B_ROWS * C_DIM;         // 8 MB
    float*    part    = (float*)(fs_h + (size_t)N_COLS * C_DIM);  // 1 KB

    const int total4 = (B_ROWS * C_DIM + N_COLS * C_DIM) / 4;
    cvt_f32_to_f16_kernel<<<(total4 + 255) / 256, 256, 0, stream>>>(
        feats, feats_s, feats_h, fs_h);

    lp_loss_kernel<<<P_IDS_, NWAVES * 32, 0, stream>>>(feats_h, fs_h, labels, part);

    finalize_kernel<<<1, P_IDS_, 0, stream>>>(part, out);
}
